// LlamaTransformerBlock_12043088298097
// MI455X (gfx1250) — matrix operand order repaired and checked by reading, not compiled, not run on hardware
//
#include <hip/hip_runtime.h>
#include <math.h>

// ---------------- problem constants ----------------
#define Bc   2
#define Sc   2048
#define Dc   2048
#define Hc   16
#define HDc  128
#define FFc  8192
#define BSc  (Bc * Sc)     // 4096 rows

typedef __attribute__((ext_vector_type(16))) __bf16 v16bf;
typedef __attribute__((ext_vector_type(8)))  float  v8f;
typedef __attribute__((ext_vector_type(4)))  float  f32x4;
typedef __attribute__((ext_vector_type(4)))  int    i32x4;

typedef __attribute__((address_space(1))) i32x4* as1_i32x4_p;  // global
typedef __attribute__((address_space(3))) i32x4* as3_i32x4_p;  // LDS

union ABfrag { v16bf v; f32x4 q[2]; };

static __device__ __forceinline__ v8f wmma_bf16(const ABfrag& a, const ABfrag& b, v8f c) {
  // D = A(16x32 bf16) * B(32x16 bf16) + C(f32)
  return __builtin_amdgcn_wmma_f32_16x16x32_bf16(false, a.v, false, b.v, (short)0, c, false, false);
}

// ---------------------------------------------------------------------------
// CDNA5 async global->LDS copy (16B per lane), tracked by ASYNCcnt.
// Falls back to a synchronous VGPR-bounce copy if the builtin is absent.
// ---------------------------------------------------------------------------
#if __has_builtin(__builtin_amdgcn_global_load_async_to_lds_b128)
#define USE_ASYNC_LDS 1
#else
#define USE_ASYNC_LDS 0
#endif

static __device__ __forceinline__ void cp16(void* lds, const void* g) {
#if USE_ASYNC_LDS
  __builtin_amdgcn_global_load_async_to_lds_b128(
      (as1_i32x4_p)(void*)g, (as3_i32x4_p)lds, 0, 0);
#else
  *(f32x4*)lds = *(const f32x4*)g;
#endif
}

static __device__ __forceinline__ void cp_wait() {
#if USE_ASYNC_LDS
#if __has_builtin(__builtin_amdgcn_s_wait_asynccnt)
  __builtin_amdgcn_s_wait_asynccnt(0);
#else
  asm volatile("s_wait_asynccnt 0x0" ::: "memory");
#endif
#endif
}

// ---------------------------------------------------------------------------
// Weight convert + transpose: src fp32 [R,C] row-major -> dst bf16 [C,R]
// ---------------------------------------------------------------------------
__global__ __launch_bounds__(256)
void wconv_kernel(const float* __restrict__ src, __bf16* __restrict__ dst, int R, int C) {
  __shared__ float tile[32][33];
  const int c0 = blockIdx.x * 32, r0 = blockIdx.y * 32;
  const int tx = threadIdx.x & 31, ty = threadIdx.x >> 5;   // 32 x 8
#pragma unroll
  for (int j = 0; j < 32; j += 8)
    tile[ty + j][tx] = src[(size_t)(r0 + ty + j) * C + c0 + tx];
  __syncthreads();
#pragma unroll
  for (int j = 0; j < 32; j += 8)
    dst[(size_t)(c0 + ty + j) * R + r0 + tx] = (__bf16)tile[tx][ty + j];
}

// ---------------------------------------------------------------------------
// RMSNorm: x fp32 [rows, Dc] -> out bf16, one block per row
// ---------------------------------------------------------------------------
__global__ __launch_bounds__(256)
void rmsnorm_kernel(const float* __restrict__ x, const float* __restrict__ g,
                    __bf16* __restrict__ out) {
  __shared__ float red[256];
  const int tid = threadIdx.x;
  const size_t row = blockIdx.x;
  const float* xr = x + row * Dc;
  float v[8];
  float ss = 0.f;
#pragma unroll
  for (int i = 0; i < 8; ++i) { v[i] = xr[tid + i * 256]; ss += v[i] * v[i]; }
  red[tid] = ss;
  __syncthreads();
  for (int s2 = 128; s2 > 0; s2 >>= 1) {
    if (tid < s2) red[tid] += red[tid + s2];
    __syncthreads();
  }
  const float r = rsqrtf(red[0] / (float)Dc + 1e-6f);
#pragma unroll
  for (int i = 0; i < 8; ++i)
    out[row * Dc + tid + i * 256] = (__bf16)(v[i] * r * g[tid + i * 256]);
}

// ---------------------------------------------------------------------------
// GEMM: C[M,N] = A[M,K](bf16,row-major) * Wt[N,K](bf16, N-major/K-contig)
// block tile 128x128, K-tile 32, 8 waves (4x2), wave tile 32x64 (2x4 frags).
// Double-buffered LDS; tile t+1 streamed with async global->LDS while tile t
// is consumed by WMMA. One barrier per K-step.
// MODE 0: store bf16 C
// MODE 1: store fp32 (C + resid)
// MODE 2: store bf16 silu(aux) * C     (SwiGLU: aux = gate, C = up)
// ---------------------------------------------------------------------------
template <int MODE>
__global__ __launch_bounds__(256)
void gemm_bf16_kernel(const __bf16* __restrict__ A, const __bf16* __restrict__ Wt,
                      void* __restrict__ outp, const float* __restrict__ resid,
                      const __bf16* __restrict__ aux, int M, int N, int K) {
  constexpr int LDT = 40;                       // 32 + 8 pad bf16 (80B rows, 16B aligned)
  __shared__ __bf16 As[2][128 * LDT];
  __shared__ __bf16 Bs[2][128 * LDT];
  const int tid = threadIdx.x;
  const int lane = tid & 31, wave = tid >> 5;
  const int half = lane >> 4, l16 = lane & 15;
  const int wm = wave >> 1, wn = wave & 1;      // 4 x 2 wave grid
  const int m0 = blockIdx.y * 128, n0 = blockIdx.x * 128;

  // staging addresses for this thread (2 x 16B chunks of A, 2 of B per tile)
  const int c0 = tid, c1 = tid + 256;
  const int ar0 = c0 >> 2, ac0 = c0 & 3, ar1 = c1 >> 2, ac1 = c1 & 3;

  v8f acc[2][4];
  const v8f z = {};
#pragma unroll
  for (int i = 0; i < 2; ++i)
#pragma unroll
    for (int j = 0; j < 4; ++j) acc[i][j] = z;

  auto stage = [&](int k0, int buf) {
    cp16(&As[buf][ar0 * LDT + ac0 * 8], A  + (size_t)(m0 + ar0) * K + k0 + ac0 * 8);
    cp16(&As[buf][ar1 * LDT + ac1 * 8], A  + (size_t)(m0 + ar1) * K + k0 + ac1 * 8);
    cp16(&Bs[buf][ar0 * LDT + ac0 * 8], Wt + (size_t)(n0 + ar0) * K + k0 + ac0 * 8);
    cp16(&Bs[buf][ar1 * LDT + ac1 * 8], Wt + (size_t)(n0 + ar1) * K + k0 + ac1 * 8);
  };

  stage(0, 0);
  cp_wait();
  __syncthreads();

  const int T = K >> 5;
  for (int t = 0; t < T; ++t) {
    const int cur = t & 1;
    if (t + 1 < T) stage((t + 1) << 5, cur ^ 1);   // prefetch next tile into other buffer

    ABfrag af[2], bfr[4];
#pragma unroll
    for (int mf = 0; mf < 2; ++mf) {   // A frag: lane=M, K 0-7/8-15 by lane half, +16..31
      const __bf16* p = &As[cur][(wm * 32 + mf * 16 + l16) * LDT + half * 8];
      af[mf].q[0] = *(const f32x4*)p;
      af[mf].q[1] = *(const f32x4*)(p + 16);
    }
#pragma unroll
    for (int nf = 0; nf < 4; ++nf) {   // B frag: lane=N, K 0-7/8-15 by lane half, +16..31
      const __bf16* p = &Bs[cur][(wn * 64 + nf * 16 + l16) * LDT + half * 8];
      bfr[nf].q[0] = *(const f32x4*)p;
      bfr[nf].q[1] = *(const f32x4*)(p + 16);
    }
#pragma unroll
    for (int mf = 0; mf < 2; ++mf)
#pragma unroll
      for (int nf = 0; nf < 4; ++nf)
        acc[mf][nf] = wmma_bf16(af[mf], bfr[nf], acc[mf][nf]);

    if (t + 1 < T) cp_wait();          // landing of tile t+1 (this wave's issues)
    __syncthreads();                   // all waves done reading cur / writing nxt
  }

  // epilogue: C/D layout -> M = vgpr + 8*(lane>=16), N = lane&15
#pragma unroll
  for (int mf = 0; mf < 2; ++mf)
#pragma unroll
    for (int nf = 0; nf < 4; ++nf)
#pragma unroll
      for (int vg = 0; vg < 8; ++vg) {
        const int gm = m0 + wm * 32 + mf * 16 + vg + half * 8;
        const int gn = n0 + wn * 64 + nf * 16 + l16;
        const size_t idx = (size_t)gm * N + gn;
        const float c = acc[mf][nf][vg];
        if (MODE == 0) {
          ((__bf16*)outp)[idx] = (__bf16)c;
        } else if (MODE == 1) {
          ((float*)outp)[idx] = c + resid[idx];
        } else {
          const float g = (float)aux[idx];
          ((__bf16*)outp)[idx] = (__bf16)((g / (1.f + __expf(-g))) * c);
        }
      }
}

// ---------------------------------------------------------------------------
// RoPE + head transpose: in bf16 [B,S,H,HD] -> out bf16 [B,H,S,HD]
// ---------------------------------------------------------------------------
__global__ __launch_bounds__(256)
void rope_kernel(const __bf16* __restrict__ in, __bf16* __restrict__ out) {
  const size_t idx = (size_t)blockIdx.x * 256 + threadIdx.x;  // B*S*H*64
  const int i = (int)(idx & 63);
  size_t t = idx >> 6;
  const int h = (int)(t % Hc); t /= Hc;
  const int s = (int)(t % Sc); const int b = (int)(t / Sc);
  const size_t src = ((size_t)(b * Sc + s) * Hc + h) * HDc;
  const float x1 = (float)in[src + i];
  const float x2 = (float)in[src + i + 64];
  const float ang = (float)s * powf(10000.f, -2.f * (float)i / 128.f);
  const float cs = cosf(ang), sn = sinf(ang);
  const size_t dst = ((size_t)(b * Hc + h) * Sc + s) * HDc;
  out[dst + i]      = (__bf16)(x1 * cs - x2 * sn);
  out[dst + i + 64] = (__bf16)(x2 * cs + x1 * sn);
}

// V transpose: in bf16 [B,S,H,HD] -> out bf16 [B,H,HD,S]
__global__ __launch_bounds__(256)
void vtrans_kernel(const __bf16* __restrict__ in, __bf16* __restrict__ out) {
  const size_t idx = (size_t)blockIdx.x * 256 + threadIdx.x;  // B*S*D
  const int hd = (int)(idx & (HDc - 1));
  size_t t = idx >> 7;
  const int h = (int)(t % Hc); t /= Hc;
  const int s = (int)(t % Sc); const int b = (int)(t / Sc);
  out[((size_t)(b * Hc + h) * HDc + hd) * Sc + s] = in[idx];
}

// ---------------------------------------------------------------------------
// Flash attention: block = (b, h, 128-row q tile), 256 thr (8 waves).
// q,k: bf16 [B,H,S,HD]; v: bf16 [B,H,HD,S]; ctx: bf16 [B,S,D] (head-concat)
// ---------------------------------------------------------------------------
#define SM_QT   0u          // 128*136 bf16 = 34816
#define SM_KT   34816u      //  64*136 bf16 = 17408
#define SM_VT   52224u      // 128* 72 bf16 = 18432
#define SM_ST   70656u      // 128* 68 f32  = 34816
#define SM_PT   105472u     // 128* 72 bf16 = 18432
#define SM_M    123904u
#define SM_L    124416u
#define SM_SC   124928u
#define SM_TOTAL 125440u

__global__ __launch_bounds__(256)
void attn_kernel(const __bf16* __restrict__ qr, const __bf16* __restrict__ kr,
                 const __bf16* __restrict__ vt, __bf16* __restrict__ ctx) {
  extern __shared__ char smem[];
  __bf16* Qt = (__bf16*)(smem + SM_QT);
  __bf16* Kt = (__bf16*)(smem + SM_KT);
  __bf16* Vt = (__bf16*)(smem + SM_VT);
  float*  St = (float*)(smem + SM_ST);
  __bf16* Pt = (__bf16*)(smem + SM_PT);
  float* mrow = (float*)(smem + SM_M);
  float* lrow = (float*)(smem + SM_L);
  float* srow = (float*)(smem + SM_SC);

  const int tid = threadIdx.x;
  const int lane = tid & 31, wave = tid >> 5;
  const int half = lane >> 4, l16 = lane & 15;
  const int wm = wave >> 1, wn = wave & 1;
  const int q0 = blockIdx.x * 128;
  const int h = blockIdx.y, b = blockIdx.z;
  const float SCALE = 0.08838834764831845f;   // 1/sqrt(128)

  const __bf16* Qg = qr + ((size_t)(b * Hc + h) * Sc + q0) * HDc;
  const __bf16* Kg = kr + ((size_t)(b * Hc + h) * Sc) * HDc;
  const __bf16* Vg = vt + ((size_t)(b * Hc + h) * HDc) * Sc;

  // stage Q tile [128 x 128] once (2048 16B chunks, async)
#pragma unroll
  for (int i = 0; i < 8; ++i) {
    const int c = tid + i * 256, row = c >> 4, cc = c & 15;
    cp16(&Qt[row * 136 + cc * 8], Qg + (size_t)row * HDc + cc * 8);
  }
  if (tid < 128) { mrow[tid] = -3.0e38f; lrow[tid] = 0.f; }

  v8f o[2][4];
  const v8f z = {};
#pragma unroll
  for (int i = 0; i < 2; ++i)
#pragma unroll
    for (int j = 0; j < 4; ++j) o[i][j] = z;
  __syncthreads();

  const int tmax = (q0 + 127) >> 6;
  for (int t = 0; t <= tmax; ++t) {
    const int kv0 = t * 64;
    // stage K tile [64 x 128] and V tile [128(hd) x 64(kv)] (async)
#pragma unroll
    for (int i = 0; i < 4; ++i) {
      const int c = tid + i * 256;
      const int krw = c >> 4, kcc = c & 15;
      cp16(&Kt[krw * 136 + kcc * 8], Kg + (size_t)(kv0 + krw) * HDc + kcc * 8);
      const int vrw = c >> 3, vcc = c & 7;
      cp16(&Vt[vrw * 72 + vcc * 8], Vg + (size_t)vrw * Sc + kv0 + vcc * 8);
    }
    cp_wait();
    __syncthreads();

    // scores S[128x64] = Q * K^T ; wave tile 32x32 (waves 4x2)
    v8f sa[2][2];
#pragma unroll
    for (int i = 0; i < 2; ++i)
#pragma unroll
      for (int j = 0; j < 2; ++j) sa[i][j] = z;
#pragma unroll
    for (int kk = 0; kk < 128; kk += 32) {
      ABfrag af[2], bfr[2];
#pragma unroll
      for (int mf = 0; mf < 2; ++mf) {
        const __bf16* p = &Qt[(wm * 32 + mf * 16 + l16) * 136 + kk + half * 8];
        af[mf].q[0] = *(const f32x4*)p;
        af[mf].q[1] = *(const f32x4*)(p + 16);
      }
#pragma unroll
      for (int nf = 0; nf < 2; ++nf) {
        const __bf16* p = &Kt[(wn * 32 + nf * 16 + l16) * 136 + kk + half * 8];
        bfr[nf].q[0] = *(const f32x4*)p;
        bfr[nf].q[1] = *(const f32x4*)(p + 16);
      }
#pragma unroll
      for (int mf = 0; mf < 2; ++mf)
#pragma unroll
        for (int nf = 0; nf < 2; ++nf)
          sa[mf][nf] = wmma_bf16(af[mf], bfr[nf], sa[mf][nf]);
    }
#pragma unroll
    for (int mf = 0; mf < 2; ++mf)
#pragma unroll
      for (int nf = 0; nf < 2; ++nf)
#pragma unroll
        for (int vg = 0; vg < 8; ++vg)
          St[(wm * 32 + mf * 16 + vg + half * 8) * 68 + wn * 32 + nf * 16 + l16] =
              sa[mf][nf][vg];
    __syncthreads();

    // online softmax, one thread per q row
    if (tid < 128) {
      const int r = tid, gq = q0 + r;
      const float mold = mrow[r];
      float mx = mold;
      for (int c = 0; c < 64; ++c)
        if (kv0 + c <= gq) mx = fmaxf(mx, St[r * 68 + c] * SCALE);
      const float scl = __expf(mold - mx);
      float lsum = 0.f;
      for (int c = 0; c < 64; ++c) {
        float p = 0.f;
        if (kv0 + c <= gq) p = __expf(St[r * 68 + c] * SCALE - mx);
        Pt[r * 72 + c] = (__bf16)p;
        lsum += p;
      }
      mrow[r] = mx;
      lrow[r] = lrow[r] * scl + lsum;
      srow[r] = scl;
    }
    __syncthreads();

    // rescale O, then O += P * V^T ; wave tile 32x64 (waves 4x2)
#pragma unroll
    for (int mf = 0; mf < 2; ++mf)
#pragma unroll
      for (int nf = 0; nf < 4; ++nf)
#pragma unroll
        for (int vg = 0; vg < 8; ++vg)
          o[mf][nf][vg] *= srow[wm * 32 + mf * 16 + vg + half * 8];
#pragma unroll
    for (int kk = 0; kk < 64; kk += 32) {
      ABfrag af[2], bfr[4];
#pragma unroll
      for (int mf = 0; mf < 2; ++mf) {
        const __bf16* p = &Pt[(wm * 32 + mf * 16 + l16) * 72 + kk + half * 8];
        af[mf].q[0] = *(const f32x4*)p;
        af[mf].q[1] = *(const f32x4*)(p + 16);
      }
#pragma unroll
      for (int nf = 0; nf < 4; ++nf) {
        const __bf16* p = &Vt[(wn * 64 + nf * 16 + l16) * 72 + kk + half * 8];
        bfr[nf].q[0] = *(const f32x4*)p;
        bfr[nf].q[1] = *(const f32x4*)(p + 16);
      }
#pragma unroll
      for (int mf = 0; mf < 2; ++mf)
#pragma unroll
        for (int nf = 0; nf < 4; ++nf)
          o[mf][nf] = wmma_bf16(af[mf], bfr[nf], o[mf][nf]);
    }
    __syncthreads();
  }

  // epilogue: divide by l, write ctx bf16 [B,S,D] head-concat
#pragma unroll
  for (int mf = 0; mf < 2; ++mf)
#pragma unroll
    for (int nf = 0; nf < 4; ++nf)
#pragma unroll
      for (int vg = 0; vg < 8; ++vg) {
        const int r = wm * 32 + mf * 16 + vg + half * 8;
        const int col = wn * 64 + nf * 16 + l16;
        const float val = o[mf][nf][vg] / lrow[r];
        ctx[((size_t)(b * Sc) + q0 + r) * Dc + h * HDc + col] = (__bf16)val;
      }
}

// ---------------------------------------------------------------------------
extern "C" void kernel_launch(void* const* d_in, const int* in_sizes, int n_in,
                              void* d_out, int out_size, void* d_ws, size_t ws_size,
                              hipStream_t stream) {
  (void)in_sizes; (void)n_in; (void)out_size; (void)ws_size;
  const float* x  = (const float*)d_in[0];
  const float* g1 = (const float*)d_in[1];
  const float* g2 = (const float*)d_in[2];
  const float* wq = (const float*)d_in[3];
  const float* wk = (const float*)d_in[4];
  const float* wv = (const float*)d_in[5];
  const float* wo = (const float*)d_in[6];
  const float* wg = (const float*)d_in[7];
  const float* wu = (const float*)d_in[8];
  const float* wd = (const float*)d_in[9];

  char* ws = (char*)d_ws;
  size_t off = 0;
  auto alloc = [&](size_t bytes) -> char* {
    char* p = ws + off;
    off += (bytes + 255) & ~(size_t)255;
    return p;
  };
  __bf16* wqt  = (__bf16*)alloc((size_t)Dc * Dc * 2);
  __bf16* wkt  = (__bf16*)alloc((size_t)Dc * Dc * 2);
  __bf16* wvt  = (__bf16*)alloc((size_t)Dc * Dc * 2);
  __bf16* wot  = (__bf16*)alloc((size_t)Dc * Dc * 2);
  __bf16* wgt  = (__bf16*)alloc((size_t)Dc * FFc * 2);
  __bf16* wut  = (__bf16*)alloc((size_t)Dc * FFc * 2);
  __bf16* wdt  = (__bf16*)alloc((size_t)FFc * Dc * 2);
  __bf16* xn   = (__bf16*)alloc((size_t)BSc * Dc * 2);
  __bf16* qtm  = (__bf16*)alloc((size_t)BSc * Dc * 2);
  __bf16* ktm  = (__bf16*)alloc((size_t)BSc * Dc * 2);
  __bf16* vtm  = (__bf16*)alloc((size_t)BSc * Dc * 2);
  __bf16* qrot = (__bf16*)alloc((size_t)BSc * Dc * 2);
  __bf16* krot = (__bf16*)alloc((size_t)BSc * Dc * 2);
  __bf16* vtr  = (__bf16*)alloc((size_t)BSc * Dc * 2);
  __bf16* ctx  = (__bf16*)alloc((size_t)BSc * Dc * 2);
  float*  hbuf = (float*) alloc((size_t)BSc * Dc * 4);
  __bf16* hn   = (__bf16*)alloc((size_t)BSc * Dc * 2);
  __bf16* gbf  = (__bf16*)alloc((size_t)BSc * FFc * 2);
  __bf16* ffin = (__bf16*)alloc((size_t)BSc * FFc * 2);

  // 1. weight convert + transpose (fp32 [K,N] -> bf16 [N,K])
  wconv_kernel<<<dim3(Dc / 32, Dc / 32), 256, 0, stream>>>(wq, wqt, Dc, Dc);
  wconv_kernel<<<dim3(Dc / 32, Dc / 32), 256, 0, stream>>>(wk, wkt, Dc, Dc);
  wconv_kernel<<<dim3(Dc / 32, Dc / 32), 256, 0, stream>>>(wv, wvt, Dc, Dc);
  wconv_kernel<<<dim3(Dc / 32, Dc / 32), 256, 0, stream>>>(wo, wot, Dc, Dc);
  wconv_kernel<<<dim3(FFc / 32, Dc / 32), 256, 0, stream>>>(wg, wgt, Dc, FFc);
  wconv_kernel<<<dim3(FFc / 32, Dc / 32), 256, 0, stream>>>(wu, wut, Dc, FFc);
  wconv_kernel<<<dim3(Dc / 32, FFc / 32), 256, 0, stream>>>(wd, wdt, FFc, Dc);

  // 2. xn = rmsnorm(x, g1)
  rmsnorm_kernel<<<BSc, 256, 0, stream>>>(x, g1, xn);

  // 3. Q/K/V projections (bf16 WMMA GEMMs)
  gemm_bf16_kernel<0><<<dim3(Dc / 128, BSc / 128), 256, 0, stream>>>(
      xn, wqt, (void*)qtm, nullptr, nullptr, BSc, Dc, Dc);
  gemm_bf16_kernel<0><<<dim3(Dc / 128, BSc / 128), 256, 0, stream>>>(
      xn, wkt, (void*)ktm, nullptr, nullptr, BSc, Dc, Dc);
  gemm_bf16_kernel<0><<<dim3(Dc / 128, BSc / 128), 256, 0, stream>>>(
      xn, wvt, (void*)vtm, nullptr, nullptr, BSc, Dc, Dc);

  // 4. RoPE + layout transforms
  rope_kernel<<<(BSc * Hc * 64) / 256, 256, 0, stream>>>(qtm, qrot);
  rope_kernel<<<(BSc * Hc * 64) / 256, 256, 0, stream>>>(ktm, krot);
  vtrans_kernel<<<((size_t)BSc * Dc) / 256, 256, 0, stream>>>(vtm, vtr);

  // 5. flash attention
  attn_kernel<<<dim3(Sc / 128, Hc, Bc), 256, SM_TOTAL, stream>>>(qrot, krot, vtr, ctx);

  // 6. h = x + ctx @ wo
  gemm_bf16_kernel<1><<<dim3(Dc / 128, BSc / 128), 256, 0, stream>>>(
      ctx, wot, (void*)hbuf, x, nullptr, BSc, Dc, Dc);

  // 7. hn = rmsnorm(h, g2)
  rmsnorm_kernel<<<BSc, 256, 0, stream>>>(hbuf, g2, hn);

  // 8. SwiGLU: gate, then up fused with silu(gate)*up
  gemm_bf16_kernel<0><<<dim3(FFc / 128, BSc / 128), 256, 0, stream>>>(
      hn, wgt, (void*)gbf, nullptr, nullptr, BSc, FFc, Dc);
  gemm_bf16_kernel<2><<<dim3(FFc / 128, BSc / 128), 256, 0, stream>>>(
      hn, wut, (void*)ffin, nullptr, gbf, BSc, FFc, Dc);

  // 9. out = h + ffin @ w_down   (fp32 output)
  gemm_bf16_kernel<1><<<dim3(Dc / 128, BSc / 128), 256, 0, stream>>>(
      ffin, wdt, d_out, hbuf, nullptr, BSc, Dc, FFc);
}
